// Bert_BiLSTM_13391708029215
// MI455X (gfx1250) — matrix operand order repaired and checked by reading, not compiled, not run on hardware
//
#include <hip/hip_runtime.h>
#include <hip/hip_bf16.h>

typedef __bf16 bf16;
typedef __attribute__((ext_vector_type(16))) __bf16 v16bf;
typedef __attribute__((ext_vector_type(8)))  __bf16 v8bf;
typedef __attribute__((ext_vector_type(8)))  float  v8f;

union V16 { v16bf v; v8bf h[2]; };

__device__ __forceinline__ v8f wmma_bf16(v16bf a, v16bf b, v8f c) {
  return __builtin_amdgcn_wmma_f32_16x16x32_bf16(false, a, false, b, (short)0, c, false, false);
}

// CDNA5 async global->LDS copy (ASYNCcnt-tracked, no VGPR round trip).
// One offset immediate applies to BOTH the LDS and global address (ISA 08_async_tensor §4.4).
__device__ __forceinline__ void async_stage_32B(unsigned lds_byte_addr, const bf16* gsrc) {
  asm volatile(
      "global_load_async_to_lds_b128 %0, %1, off\n\t"
      "global_load_async_to_lds_b128 %0, %1, off offset:16"
      :: "v"(lds_byte_addr), "v"(gsrc) : "memory");
}
__device__ __forceinline__ void wait_async0() {
  asm volatile("s_wait_asynccnt 0x0" ::: "memory");
}

// ---------------- 1) wordpiece -> word mean pooling (bf16 out) ----------------
// word_ids are sorted per row: run-length scan, no atomics.
__global__ void __launch_bounds__(256) pool_kernel(const float* __restrict__ hs,
                                                   const int* __restrict__ wid,
                                                   bf16* __restrict__ pooled) {
  const int b = blockIdx.x;
  const int d = blockIdx.y * 256 + threadIdx.x;   // D=768, gridDim.y=3
  for (int w = 0; w < 256; ++w)
    pooled[((size_t)b * 256 + w) * 768 + d] = (bf16)0.0f;
  float sum = 0.0f, cnt = 0.0f;
  int cur = wid[(size_t)b * 512];
  for (int t = 0; t < 512; ++t) {
    int w = wid[(size_t)b * 512 + t];          // uniform across block -> scalar branch
    if (w != cur) {
      pooled[((size_t)b * 256 + cur) * 768 + d] = (bf16)(sum / cnt);
      sum = 0.0f; cnt = 0.0f; cur = w;
    }
    sum += hs[((size_t)b * 512 + t) * 768 + d];
    cnt += 1.0f;
  }
  pooled[((size_t)b * 256 + cur) * 768 + d] = (bf16)(sum / cnt);
}

// ---------------- 2) repack weights f32 -> bf16 WMMA B-fragment layout --------
// dst[((nt*KT + kt)*32 + lane)*16 + j] = w[k*1024 + n]
//   n = nt*16 + (lane&15);  k = kt*32 + (lane&16 ? 16 : 0) + j   (B 32x16 layout)
__global__ void __launch_bounds__(256) repack_kernel(const float* __restrict__ w,
                                                     bf16* __restrict__ dst, int KT) {
  int idx  = blockIdx.x * 256 + threadIdx.x;
  int j    = idx & 15;
  int lane = (idx >> 4) & 31;
  int f    = idx >> 9;            // frag index = nt*KT + kt
  int kt   = f % KT;
  int nt   = f / KT;
  if (nt >= 64) return;
  int n = nt * 16 + (lane & 15);
  int k = kt * 32 + ((lane & 16) ? 16 : 0) + j;
  dst[idx] = (bf16)w[(size_t)k * 1024 + n];
}

// ---------------- 3) input projection: pre = pooled @ W_ih + b (bf16 out) -----
// grid (128, 8, 2): 8 waves/block, wave owns one 16-row M-tile; 8 N-tiles/block.
// B tiles double-buffered in LDS via CDNA5 async global->LDS DMA.
__global__ void __launch_bounds__(256) gemm_ih(const bf16* __restrict__ pooled,
                                               const bf16* __restrict__ wihf,
                                               const bf16* __restrict__ wihb,
                                               const float* __restrict__ bfd,
                                               const float* __restrict__ bbd,
                                               bf16* __restrict__ pre) {
  const int dir = blockIdx.z;
  const bf16* wfrag = dir ? wihb : wihf;
  const float* bias = dir ? bbd : bfd;
  bf16* pre_d = pre + (size_t)dir * 16384 * 1024;

  __shared__ __align__(16) bf16 bstage[2 * 8 * 32 * 16];   // 16 KB, double buffered

  const int tid  = threadIdx.x;
  const int wave = tid >> 5, lane = tid & 31;
  const int mlo  = lane & 15;
  const int mt   = blockIdx.x * 8 + wave;   // 0..1023
  const int n0   = blockIdx.y * 128;

  // LDS byte address of this thread's 32B staging slot (low 32 bits of flat addr)
  const unsigned lds_base = (unsigned)(uintptr_t)(&bstage[0]) + (unsigned)(wave * 32 + lane) * 32u;

  v8f acc[8];
#pragma unroll
  for (int i = 0; i < 8; ++i) {
    float bv = bias[n0 + i * 16 + mlo];
#pragma unroll
    for (int r = 0; r < 8; ++r) acc[i][r] = bv;
  }

  const int arow = mt * 16 + mlo;
  const int ntg  = blockIdx.y * 8 + wave;       // frag this wave stages

  // prologue: async-stage kt=0 into buffer 0
  async_stage_32B(lds_base, wfrag + ((size_t)(ntg * 24 + 0) * 32 + lane) * 16);
  wait_async0();
  __syncthreads();

  for (int kt = 0; kt < 24; ++kt) {
    const int cur = kt & 1, nxt = cur ^ 1;
    if (kt + 1 < 24)     // async DMA of next K-step overlaps this step's WMMAs
      async_stage_32B(lds_base + (unsigned)nxt * 8192u,
                      wfrag + ((size_t)(ntg * 24 + kt + 1) * 32 + lane) * 16);
    // A fragment (contiguous K: lane<16 -> K 0..15, lane>=16 -> K 16..31)
    const int kb = kt * 32 + ((lane & 16) ? 16 : 0);
    const v8bf* ap = (const v8bf*)(pooled + (size_t)arow * 768 + kb);
    V16 a; a.h[0] = ap[0]; a.h[1] = ap[1];
    // load all 8 B fragments, then 8 independent WMMAs
    V16 bfr[8];
#pragma unroll
    for (int i = 0; i < 8; ++i) {
      const v8bf* bp = (const v8bf*)(bstage + (cur * 4096) + (i * 32 + lane) * 16);
      bfr[i].h[0] = bp[0]; bfr[i].h[1] = bp[1];
    }
#pragma unroll
    for (int i = 0; i < 8; ++i) acc[i] = wmma_bf16(a.v, bfr[i].v, acc[i]);
    wait_async0();
    __syncthreads();
  }

  const int hi8 = (lane & 16) ? 8 : 0;
#pragma unroll
  for (int i = 0; i < 8; ++i) {
    int n = n0 + i * 16 + mlo;
#pragma unroll
    for (int r = 0; r < 8; ++r) {
      int row = mt * 16 + r + hi8;
      pre_d[(size_t)row * 1024 + n] = (bf16)acc[i][r];
    }
  }
}

// ---------------- 4) bidirectional LSTM recurrence ----------------------------
// grid (4,2): batch-group of 16 rows x direction; 8 waves/block; 256 time steps.
// gates = h @ W_hh via WMMA (pre[t] added in the coalesced elementwise phase).
__global__ void __launch_bounds__(256) lstm_rec(const bf16* __restrict__ pre,
                                                const bf16* __restrict__ whhf,
                                                const bf16* __restrict__ whhb,
                                                float* __restrict__ out) {
  const int g   = blockIdx.x;
  const int dir = blockIdx.y;
  const bf16* wfrag = dir ? whhb : whhf;
  const bf16* pre_d = pre + (size_t)dir * 16384 * 1024;
  const int b0 = g * 16;

  __shared__ __align__(16) float gates[16 * 1024];   // 64 KB
  __shared__ __align__(16) bf16  hbuf[16 * 256];     // 8 KB

  const int tid  = threadIdx.x;
  const int wave = tid >> 5, lane = tid & 31;
  const int mlo  = lane & 15;
  const int hi8  = (lane & 16) ? 8 : 0;

  for (int i = tid; i < 16 * 256; i += 256) hbuf[i] = (bf16)0.0f;
  float c[16];
#pragma unroll
  for (int m = 0; m < 16; ++m) c[m] = 0.0f;
  __syncthreads();

  for (int s = 0; s < 256; ++s) {
    const int t = dir ? (255 - s) : s;

    // prefetch next step's pre tile (16 rows x 2KB) into cache
    if (s + 1 < 256) {
      const int tn = dir ? (254 - s) : (s + 1);
      const bf16* pp = pre_d + ((size_t)(b0 + (tid >> 4)) * 256 + tn) * 1024
                             + (size_t)(tid & 15) * 64;
      __builtin_prefetch((const void*)pp, 0, 1);
    }

    // preload all 8 A fragments (h) from LDS once per step
    V16 afr[8];
#pragma unroll
    for (int kt = 0; kt < 8; ++kt) {
      const int kb = kt * 32 + ((lane & 16) ? 16 : 0);
      const v8bf* ap = (const v8bf*)(hbuf + mlo * 256 + kb);
      afr[kt].h[0] = ap[0]; afr[kt].h[1] = ap[1];
    }

#pragma unroll
    for (int i = 0; i < 8; ++i) {
      const int nt = wave * 8 + i;
      V16 bfr[8];
#pragma unroll
      for (int kt = 0; kt < 8; ++kt) {
        const v8bf* bp = (const v8bf*)(wfrag + ((size_t)(nt * 8 + kt) * 32 + lane) * 16);
        bfr[kt].h[0] = bp[0]; bfr[kt].h[1] = bp[1];
      }
      v8f acc;
#pragma unroll
      for (int r = 0; r < 8; ++r) acc[r] = 0.0f;
#pragma unroll
      for (int kt = 0; kt < 8; ++kt) acc = wmma_bf16(afr[kt].v, bfr[kt].v, acc);
      const int n = nt * 16 + mlo;
#pragma unroll
      for (int r = 0; r < 8; ++r) gates[(r + hi8) * 1024 + n] = acc[r];
    }
    __syncthreads();

    // elementwise cell update: thread owns column n = tid for all 16 rows;
    // pre[t] added here with fully coalesced bf16 loads.
    {
      const int n = tid;
#pragma unroll 4
      for (int m = 0; m < 16; ++m) {
        const size_t pb = ((size_t)(b0 + m) * 256 + t) * 1024;
        float gi = gates[m * 1024 + n]       + (float)pre_d[pb + n];
        float gf = gates[m * 1024 + 256 + n] + (float)pre_d[pb + 256 + n];
        float gg = gates[m * 1024 + 512 + n] + (float)pre_d[pb + 512 + n];
        float go = gates[m * 1024 + 768 + n] + (float)pre_d[pb + 768 + n];
        gi = 1.0f / (1.0f + __expf(-gi));
        gf = 1.0f / (1.0f + __expf(-gf));
        gg = tanhf(gg);
        go = 1.0f / (1.0f + __expf(-go));
        c[m] = gf * c[m] + gi * gg;
        float h = go * tanhf(c[m]);
        hbuf[m * 256 + n] = (bf16)h;
        out[((size_t)(b0 + m) * 256 + t) * 512 + dir * 256 + n] = h;
      }
    }
    __syncthreads();
  }
}

// ---------------- launcher ----------------------------------------------------
extern "C" void kernel_launch(void* const* d_in, const int* in_sizes, int n_in,
                              void* d_out, int out_size, void* d_ws, size_t ws_size,
                              hipStream_t stream) {
  const float* hs     = (const float*)d_in[0];
  const float* w_ih_f = (const float*)d_in[1];
  const float* w_hh_f = (const float*)d_in[2];
  const float* b_f    = (const float*)d_in[3];
  const float* w_ih_b = (const float*)d_in[4];
  const float* w_hh_b = (const float*)d_in[5];
  const float* b_b    = (const float*)d_in[6];
  const int*   wid    = (const int*)d_in[7];
  float* out = (float*)d_out;

  char* p = (char*)d_ws;
  bf16* pooled = (bf16*)p; p += (size_t)16384 * 768 * 2;        // 25.2 MB
  bf16* wihf   = (bf16*)p; p += (size_t)64 * 24 * 512 * 2;      // 1.5 MB
  bf16* wihb   = (bf16*)p; p += (size_t)64 * 24 * 512 * 2;
  bf16* whhf   = (bf16*)p; p += (size_t)64 * 8 * 512 * 2;       // 0.5 MB
  bf16* whhb   = (bf16*)p; p += (size_t)64 * 8 * 512 * 2;
  bf16* pre    = (bf16*)p; p += (size_t)2 * 16384 * 1024 * 2;   // 67 MB

  pool_kernel<<<dim3(64, 3), 256, 0, stream>>>(hs, wid, pooled);
  repack_kernel<<<128 * 24, 256, 0, stream>>>(w_ih_f, wihf, 24);
  repack_kernel<<<128 * 24, 256, 0, stream>>>(w_ih_b, wihb, 24);
  repack_kernel<<<128 * 8, 256, 0, stream>>>(w_hh_f, whhf, 8);
  repack_kernel<<<128 * 8, 256, 0, stream>>>(w_hh_b, whhb, 8);
  gemm_ih<<<dim3(128, 8, 2), 256, 0, stream>>>(pooled, wihf, wihb, b_f, b_b, pre);
  lstm_rec<<<dim3(4, 2), 256, 0, stream>>>(pre, whhf, whhb, out);
}
